// EdgeConvResidualNodeRegressor_11888469475720
// MI455X (gfx1250) — matrix operand order repaired and checked by reading, not compiled, not run on hardware
//
#include <hip/hip_runtime.h>
#include <hip/hip_bf16.h>

// EdgeConv GNN for MI455X (gfx1250, wave32, WMMA).
// Algebraic rewrite: cat[xi, xj-xi] @ W1 = xi@(W1_top-W1_bot) + xj@W1_bot,
// so the heavy per-edge K=128 GEMM becomes per-node GEMMs (L2-resident) plus a
// per-edge 64x64 GEMM. All GEMMs use v_wmma_f32_16x16x32_f16.

#define N_NODES 50000
#define N_EDGES 800000
#define IN_DIM  128
#define HID     64
#define N_LAYERS 3

typedef __attribute__((ext_vector_type(16))) _Float16 v16h;
typedef __attribute__((ext_vector_type(8)))  _Float16 h8;
typedef __attribute__((ext_vector_type(8)))  float    v8f;
typedef __attribute__((ext_vector_type(4)))  float    f4;

#define WMMA_F16(a, b, c) \
  __builtin_amdgcn_wmma_f32_16x16x32_f16(false, (a), false, (b), (short)0, (c), false, false)

static __device__ __forceinline__ v16h hcombine(h8 lo, h8 hi) {
  return __builtin_shufflevector(lo, hi, 0,1,2,3,4,5,6,7,8,9,10,11,12,13,14,15);
}

// A-fragment (16x32 f16, MxK) from a row-major tile of halfs.
// ISA layout: lanes 0-15 = rows 0-15 with K(local) 0-7 in halfs 0-7 and
// K 16-23 in halfs 8-15; lanes 16-31 carry K 8-15 / 24-31.
static __device__ __forceinline__ v16h afrag(const _Float16* tile, int lda, int k0, int lane) {
  int r = lane & 15, hi = lane >> 4;
  const _Float16* p = tile + r * lda + k0 + hi * 8;
  h8 lo = *(const h8*)p;
  h8 hh = *(const h8*)(p + 16);
  return hcombine(lo, hh);
}

// B-fragment (32x16 f16, KxN) from a PRE-TRANSPOSED weight matrix Wt[N][K]
// (row-major, ldb halfs). Lane c holds column n0+c: halfs 0-7 = K(local)
// 0-7 / 8-15 and halfs 8-15 = K 16-23 / 24-31 (lanes 0-15 / 16-31).
static __device__ __forceinline__ v16h bfrag(const _Float16* wt, int ldb, int n0, int k0, int lane) {
  int c = lane & 15, hi = lane >> 4;
  const _Float16* p = wt + (n0 + c) * ldb + k0 + hi * 8;
  return hcombine(*(const h8*)p, *(const h8*)(p + 16));
}

static __device__ __forceinline__ h8 cvt8(const float* p) {
  f4 a = *(const f4*)p;
  f4 b = *(const f4*)(p + 4);
  h8 r;
  r[0] = (_Float16)a[0]; r[1] = (_Float16)a[1]; r[2] = (_Float16)a[2]; r[3] = (_Float16)a[3];
  r[4] = (_Float16)b[0]; r[5] = (_Float16)b[1]; r[6] = (_Float16)b[2]; r[7] = (_Float16)b[3];
  return r;
}

// Float atomic max via signed-int max / unsigned-int min trick (valid with
// -inf initialization, handles mixed signs).
static __device__ __forceinline__ void atomicMaxF(float* addr, float val) {
  if (val >= 0.0f) atomicMax((int*)addr, __float_as_int(val));
  else             atomicMin((unsigned int*)addr, __float_as_uint(val));
}

// ---------------- weight prep: transpose + f16 convert --------------------
__global__ void prep_weights(const float* __restrict__ Wp, const float* __restrict__ W1,
                             const float* __restrict__ W2, _Float16* __restrict__ WpT,
                             _Float16* __restrict__ A1T, _Float16* __restrict__ B1T,
                             _Float16* __restrict__ W2T) {
  int tid = blockIdx.x * blockDim.x + threadIdx.x;
  int stride = gridDim.x * blockDim.x;
  // WpT[n][k] = Wp[k][n]
  for (int idx = tid; idx < HID * IN_DIM; idx += stride) {
    int n = idx >> 7, k = idx & 127;
    WpT[idx] = (_Float16)Wp[k * HID + n];
  }
  // A1T[l][n][k] = W1[l][k][n] - W1[l][64+k][n];  B1T[l][n][k] = W1[l][64+k][n]
  // W2T[l][n][k] = W2[l][k][n]
  for (int idx = tid; idx < N_LAYERS * HID * HID; idx += stride) {
    int l = idx >> 12, r = idx & 4095, n = r >> 6, k = r & 63;
    float top = W1[(l * 2 * HID + k) * HID + n];
    float bot = W1[(l * 2 * HID + HID + k) * HID + n];
    A1T[idx] = (_Float16)(top - bot);
    B1T[idx] = (_Float16)bot;
    W2T[idx] = (_Float16)W2[(l * HID + k) * HID + n];
  }
}

// ---------------- input projection: h = x @ Wp + bp (WMMA, K=128) ---------
__global__ __launch_bounds__(256) void proj_kernel(const float* __restrict__ x,
                                                   const _Float16* __restrict__ WpT,
                                                   const float* __restrict__ bp,
                                                   float* __restrict__ h,
                                                   _Float16* __restrict__ hf16) {
  int wid = threadIdx.x >> 5, lane = threadIdx.x & 31;
  int tile = blockIdx.x * 8 + wid;
  if (tile >= N_NODES / 16) return;
  int m0 = tile * 16;
  int r = lane & 15, hi = lane >> 4;
  v16h a[4];
#pragma unroll
  for (int kc = 0; kc < 4; ++kc) {
    const float* p = x + (size_t)(m0 + r) * IN_DIM + kc * 32 + hi * 8;
    a[kc] = hcombine(cvt8(p), cvt8(p + 16));
  }
  int cn = lane & 15;
#pragma unroll
  for (int nt = 0; nt < 4; ++nt) {
    int n0 = nt * 16;
    v8f c = {0.f, 0.f, 0.f, 0.f, 0.f, 0.f, 0.f, 0.f};
#pragma unroll
    for (int kc = 0; kc < 4; ++kc)
      c = WMMA_F16(a[kc], bfrag(WpT, IN_DIM, n0, kc * 32, lane), c);
    float bias = bp[n0 + cn];
#pragma unroll
    for (int i = 0; i < 8; ++i) {
      int row = m0 + hi * 8 + i;
      float v = c[i] + bias;
      h[(size_t)row * HID + n0 + cn] = v;
      hf16[(size_t)row * HID + n0 + cn] = (_Float16)v;
    }
  }
}

// ---------------- per-node GEMMs: U = h@A1 + b1, V = h@B1 (WMMA) ----------
__global__ __launch_bounds__(256) void node_uv_kernel(const _Float16* __restrict__ hf16,
                                                      const _Float16* __restrict__ A1Tl,
                                                      const _Float16* __restrict__ B1Tl,
                                                      const float* __restrict__ b1l,
                                                      float* __restrict__ U,
                                                      float* __restrict__ V) {
  int wid = threadIdx.x >> 5, lane = threadIdx.x & 31;
  int tile = blockIdx.x * 8 + wid;
  if (tile >= N_NODES / 16) return;
  int m0 = tile * 16;
  v16h a0 = afrag(hf16 + (size_t)m0 * HID, HID, 0, lane);
  v16h a1 = afrag(hf16 + (size_t)m0 * HID, HID, 32, lane);
  int cn = lane & 15, hi = lane >> 4;
#pragma unroll
  for (int nt = 0; nt < 4; ++nt) {
    int n0 = nt * 16;
    v8f cu = {0.f, 0.f, 0.f, 0.f, 0.f, 0.f, 0.f, 0.f};
    cu = WMMA_F16(a0, bfrag(A1Tl, HID, n0, 0, lane), cu);
    cu = WMMA_F16(a1, bfrag(A1Tl, HID, n0, 32, lane), cu);
    v8f cv = {0.f, 0.f, 0.f, 0.f, 0.f, 0.f, 0.f, 0.f};
    cv = WMMA_F16(a0, bfrag(B1Tl, HID, n0, 0, lane), cv);
    cv = WMMA_F16(a1, bfrag(B1Tl, HID, n0, 32, lane), cv);
    float bias = b1l[n0 + cn];
#pragma unroll
    for (int i = 0; i < 8; ++i) {
      int row = m0 + hi * 8 + i;
      U[(size_t)row * HID + n0 + cn] = cu[i] + bias;
      V[(size_t)row * HID + n0 + cn] = cv[i];
    }
  }
}

// ---------------- per-edge GEMM + scatter-max (WMMA + LDS + atomics) ------
__global__ __launch_bounds__(256) void edge_kernel(const long long* __restrict__ ei,
                                                   const float* __restrict__ U,
                                                   const float* __restrict__ V,
                                                   const _Float16* __restrict__ W2Tl,
                                                   const float* __restrict__ b2l,
                                                   float* __restrict__ agg) {
  __shared__ _Float16 w2s[HID * HID];          // W2^T tile, shared by the block
  __shared__ float    b2s[HID];
  __shared__ _Float16 zt[8][16 * HID];         // per-wave Z tile (16 edges x 64)
  __shared__ int      srcs[8][16];
  __shared__ int      dsts[8][16];

  int tid = threadIdx.x;
  for (int i = tid; i < HID * HID; i += 256) w2s[i] = W2Tl[i];
  if (tid < HID) b2s[tid] = b2l[tid];

  int wid = tid >> 5, lane = tid & 31;
  int tile = blockIdx.x * 8 + wid;             // grid exact: E/16 tiles
  int e0 = tile * 16;
  if (lane < 16) {
    srcs[wid][lane] = (int)ei[e0 + lane];
    dsts[wid][lane] = (int)ei[N_EDGES + e0 + lane];
  }
  __syncthreads();

  // Z = relu(U[dst] + V[src]) -> f16 LDS tile (gathers hit L2: U,V ~12.8MB)
  for (int idx = lane; idx < 16 * HID; idx += 32) {
    int m = idx >> 6, f = idx & 63;
    float z = U[(size_t)dsts[wid][m] * HID + f] + V[(size_t)srcs[wid][m] * HID + f];
    zt[wid][idx] = (_Float16)fmaxf(z, 0.0f);
  }
  __syncthreads();

  v16h a0 = afrag(zt[wid], HID, 0, lane);
  v16h a1 = afrag(zt[wid], HID, 32, lane);
  int cn = lane & 15, hi = lane >> 4;
#pragma unroll
  for (int nt = 0; nt < 4; ++nt) {
    int n0 = nt * 16;
    v8f c = {0.f, 0.f, 0.f, 0.f, 0.f, 0.f, 0.f, 0.f};
    c = WMMA_F16(a0, bfrag(w2s, HID, n0, 0, lane), c);
    c = WMMA_F16(a1, bfrag(w2s, HID, n0, 32, lane), c);
    float bias = b2s[n0 + cn];
#pragma unroll
    for (int i = 0; i < 8; ++i) {
      int d = dsts[wid][hi * 8 + i];
      atomicMaxF(&agg[(size_t)d * HID + n0 + cn], c[i] + bias);
    }
  }
}

// ---------------- misc elementwise kernels --------------------------------
__global__ void init_agg(float* __restrict__ agg) {
  int i = blockIdx.x * blockDim.x + threadIdx.x;
  if (i < N_NODES * HID) agg[i] = __uint_as_float(0xFF800000u);  // -inf
}

__global__ void finalize_kernel(const float* __restrict__ agg, float* __restrict__ h,
                                _Float16* __restrict__ hf16) {
  int i = blockIdx.x * blockDim.x + threadIdx.x;
  if (i >= N_NODES * HID) return;
  float a = agg[i];
  unsigned u = __float_as_uint(a);
  if ((u & 0x7f800000u) == 0x7f800000u) a = 0.0f;  // non-finite (no in-edges) -> 0
  a = fmaxf(a, 0.0f);                              // relu
  float v = a + h[i];                              // residual
  h[i] = v;
  hf16[i] = (_Float16)v;
}

__global__ void out_kernel(const float* __restrict__ h, const float* __restrict__ Wo,
                           const float* __restrict__ bo, float* __restrict__ out) {
  int n = blockIdx.x * blockDim.x + threadIdx.x;
  if (n >= N_NODES) return;
  float s = bo[0];
#pragma unroll
  for (int k = 0; k < HID; ++k) s += h[(size_t)n * HID + k] * Wo[k];
  out[n] = s;
}

// ---------------- host launcher -------------------------------------------
extern "C" void kernel_launch(void* const* d_in, const int* in_sizes, int n_in,
                              void* d_out, int out_size, void* d_ws, size_t ws_size,
                              hipStream_t stream) {
  const float*     x  = (const float*)d_in[0];
  const long long* ei = (const long long*)d_in[1];  // int64 edge_index [2,E]
  const float*     Wp = (const float*)d_in[2];
  const float*     bp = (const float*)d_in[3];
  const float*     W1 = (const float*)d_in[4];
  const float*     b1 = (const float*)d_in[5];
  const float*     W2 = (const float*)d_in[6];
  const float*     b2 = (const float*)d_in[7];
  const float*     Wo = (const float*)d_in[8];
  const float*     bo = (const float*)d_in[9];
  float* out = (float*)d_out;

  char* ws = (char*)d_ws;
  constexpr size_t SZ_H   = (size_t)N_NODES * HID * sizeof(float);     // 12.8 MB
  constexpr size_t SZ_H16 = (size_t)N_NODES * HID * sizeof(_Float16);  // 6.4 MB
  float*     h    = (float*)(ws);
  _Float16*  hf16 = (_Float16*)(ws + SZ_H);
  float*     U    = (float*)(ws + SZ_H + SZ_H16);
  float*     V    = (float*)(ws + 2 * SZ_H + SZ_H16);
  float*     agg  = (float*)(ws + 3 * SZ_H + SZ_H16);
  _Float16*  WpT  = (_Float16*)(ws + 4 * SZ_H + SZ_H16);
  _Float16*  A1T  = WpT + HID * IN_DIM;
  _Float16*  B1T  = A1T + N_LAYERS * HID * HID;
  _Float16*  W2T  = B1T + N_LAYERS * HID * HID;

  const int node_tiles = N_NODES / 16;                 // 3125 (exact)
  const int node_blocks = (node_tiles + 7) / 8;        // 391
  const int edge_blocks = N_EDGES / 16 / 8;            // 6250 (exact)
  const int elem_blocks = (N_NODES * HID + 255) / 256; // 12500

  prep_weights<<<64, 256, 0, stream>>>(Wp, W1, W2, WpT, A1T, B1T, W2T);
  proj_kernel<<<node_blocks, 256, 0, stream>>>(x, WpT, bp, h, hf16);
  for (int l = 0; l < N_LAYERS; ++l) {
    init_agg<<<elem_blocks, 256, 0, stream>>>(agg);
    node_uv_kernel<<<node_blocks, 256, 0, stream>>>(hf16, A1T + l * HID * HID,
                                                    B1T + l * HID * HID, b1 + l * HID, U, V);
    edge_kernel<<<edge_blocks, 256, 0, stream>>>(ei, U, V, W2T + l * HID * HID,
                                                 b2 + l * HID, agg);
    finalize_kernel<<<elem_blocks, 256, 0, stream>>>(agg, h, hf16);
  }
  out_kernel<<<(N_NODES + 255) / 256, 256, 0, stream>>>(h, Wo, bo, out);
}
